// Attention_58703613001761
// MI455X (gfx1250) — matrix operand order repaired and checked by reading, not compiled, not run on hardware
//
#include <hip/hip_runtime.h>

// ---------------------------------------------------------------------------
// MI455X (gfx1250) attention: x@Wqkv+b -> flash attention -> @Wproj+b
// bf16 WMMA (f32 accumulate) + Tensor Data Mover tile feeds (double-buffered).
// ---------------------------------------------------------------------------

typedef __attribute__((ext_vector_type(8)))  __bf16 v8bf;
typedef __attribute__((ext_vector_type(16))) __bf16 v16bf;
typedef __attribute__((ext_vector_type(8)))  float  v8f;
typedef __attribute__((ext_vector_type(4)))  unsigned int v4u;
typedef __attribute__((ext_vector_type(8)))  int v8i;
typedef __attribute__((ext_vector_type(4)))  int v4i;

#define WMMA_BF16(A, B, C) \
  __builtin_amdgcn_wmma_f32_16x16x32_bf16(false, (A), false, (B), (short)0, (C), false, false)

#if __has_builtin(__builtin_amdgcn_tensor_load_to_lds) && \
    __has_builtin(__builtin_amdgcn_s_wait_tensorcnt)
#define HAVE_TDM 1
#else
#define HAVE_TDM 0
#endif

static __device__ inline unsigned short f2bf_raw(float f) {
  unsigned u = __builtin_bit_cast(unsigned, f);
  u += 0x7FFFu + ((u >> 16) & 1u);          // round-to-nearest-even
  return (unsigned short)(u >> 16);
}

static __device__ inline unsigned ldsoff(const void* p) {
  // generic pointers to LDS carry the byte offset in the low 32 bits
  return (unsigned)(unsigned long long)(uintptr_t)p;
}

#if HAVE_TDM
// 2D bf16 tile load global->LDS via TDM. pad_interval/pad_amount are the
// encoded D# codes: interval code c -> pad every 2^(c+1) DWORDs, amount code
// a -> insert (a+1) DWORDs. Dims/strides in elements (data_size = 2 bytes).
static __device__ inline void tdm_load_2d(unsigned lds_byte_off,
                                          const unsigned short* gp,
                                          unsigned tensor_d0, unsigned tensor_d1,
                                          unsigned stride0,
                                          unsigned tile_d0, unsigned tile_d1,
                                          unsigned pad_interval, unsigned pad_amount) {
  unsigned long long ga = (unsigned long long)(uintptr_t)gp;
  v4u g0;
  g0[0] = 1u;                                            // count=1, user mode
  g0[1] = lds_byte_off;                                  // lds_addr
  g0[2] = (unsigned)(ga & 0xFFFFFFFFu);                  // global_addr[31:0]
  g0[3] = (unsigned)((ga >> 32) & 0x1FFFFFFu) | (2u << 30); // [56:32] | type=2
  v8i g1;
  g1[0] = (int)((1u << 16) |                  // data_size = 2 bytes
                (1u << 20) |                  // pad_enable
                (pad_interval << 22) | (pad_amount << 25));
  g1[1] = (int)(tensor_d0 << 16);                         // tensor_dim0[15:0]
  g1[2] = (int)((tensor_d0 >> 16) | (tensor_d1 << 16));   // td0[31:16]|td1[15:0]
  g1[3] = (int)((tensor_d1 >> 16) | (tile_d0 << 16));     // td1[31:16]|tile0
  g1[4] = (int)(tile_d1 & 0xFFFFu);                       // tile1; tile2=0
  g1[5] = (int)stride0;                                   // dim0_stride[31:0]
  g1[6] = 0;                                              // stride hi + dim1_stride lo
  g1[7] = 0;
  v4i z4 = {0, 0, 0, 0};
  v8i z8 = {0, 0, 0, 0, 0, 0, 0, 0};
  __builtin_amdgcn_tensor_load_to_lds(g0, g1, z4, z4, z8, 0);
}
#endif

// A-fragment (16x32 bf16, MxK): lane<16 -> row=lane, K = {8h..8h+7, 16+8h..16+8h+7}
static __device__ inline v16bf load_a_frag(const unsigned short* base, int rowStride,
                                           int m0, int k0, int lane) {
  int m = m0 + (lane & 15);
  int half = lane >> 4;
  const __bf16* p = (const __bf16*)(base + (size_t)m * rowStride + k0 + 8 * half);
  v8bf lo = *(const v8bf*)p;
  v8bf hi = *(const v8bf*)(p + 16);
  v16bf r;
#pragma unroll
  for (int i = 0; i < 8; ++i) { r[i] = lo[i]; r[i + 8] = hi[i]; }
  return r;
}

// B-fragment (32x16 bf16, KxN) from an [N][K] buffer: lane -> n=lane&15,
// K = {8h..8h+7, 16+8h..16+8h+7}, h = lane>>4.
static __device__ inline v16bf load_b_frag(const unsigned short* base, int rowStride,
                                           int n0, int k0, int lane) {
  int n = n0 + (lane & 15);
  int half = lane >> 4;
  const __bf16* p = (const __bf16*)(base + (size_t)n * rowStride + k0 + 8 * half);
  v8bf lo = *(const v8bf*)p;
  v8bf hi = *(const v8bf*)(p + 16);
  v16bf r;
#pragma unroll
  for (int i = 0; i < 8; ++i) { r[i] = lo[i]; r[i + 8] = hi[i]; }
  return r;
}

// ---------------------------------------------------------------------------
// Kernel 0a: elementwise f32 -> bf16 (n multiple of 8)
// ---------------------------------------------------------------------------
__global__ __launch_bounds__(256) void cvt_bf16_kernel(
    const float* __restrict__ in, unsigned short* __restrict__ out, int n) {
  int i = (blockIdx.x * 256 + threadIdx.x) * 8;
  if (i >= n) return;
  float4 a = *(const float4*)(in + i);
  float4 b = *(const float4*)(in + i + 4);
  uint4 r;
  r.x = (unsigned)f2bf_raw(a.x) | ((unsigned)f2bf_raw(a.y) << 16);
  r.y = (unsigned)f2bf_raw(a.z) | ((unsigned)f2bf_raw(a.w) << 16);
  r.z = (unsigned)f2bf_raw(b.x) | ((unsigned)f2bf_raw(b.y) << 16);
  r.w = (unsigned)f2bf_raw(b.z) | ((unsigned)f2bf_raw(b.w) << 16);
  *(uint4*)(out + i) = r;
}

// ---------------------------------------------------------------------------
// Kernel 0b: W [1024][ncols] f32 -> Wt [ncols][1024] bf16 (64x64 LDS tiles)
// ---------------------------------------------------------------------------
__global__ __launch_bounds__(256) void cvt_transpose_kernel(
    const float* __restrict__ W, unsigned short* __restrict__ Wt, int ncols) {
  __shared__ alignas(16) unsigned short Tl[64 * 72];
  const int tid = threadIdx.x;
  const int n0 = blockIdx.x * 64, k0 = blockIdx.y * 64;
  {
    int r = tid >> 2, c0 = (tid & 3) * 16;  // r: k row, c0: n col chunk
    const float* src = W + (size_t)(k0 + r) * ncols + n0 + c0;
#pragma unroll
    for (int u = 0; u < 16; ++u) Tl[(c0 + u) * 72 + r] = f2bf_raw(src[u]);
  }
  __syncthreads();
  {
    int rn = tid >> 2, kc = (tid & 3) * 16;
    unsigned short* dst = Wt + (size_t)(n0 + rn) * 1024 + k0 + kc;
    const unsigned short* s = Tl + rn * 72 + kc;
#pragma unroll
    for (int u = 0; u < 16; ++u) dst[u] = s[u];
  }
}

// ---------------------------------------------------------------------------
// Kernel 1: qkv = Xbf @ WtQKV^T + b ; scatter Q[bh][n][d], K[bh][n][d], Vt[bh][d][n]
// Block tile 64x128, 8 waves (4x2), K-chunks of 32, TDM double-buffered tiles.
// ---------------------------------------------------------------------------
__global__ __launch_bounds__(256) void qkv_gemm_kernel(
    const unsigned short* __restrict__ Xbf, const unsigned short* __restrict__ Wt,
    const float* __restrict__ bqkv,
    unsigned short* __restrict__ Qws, unsigned short* __restrict__ Kws,
    unsigned short* __restrict__ Vtws) {
  __shared__ alignas(16) unsigned short Al[2][64 * 40];
  __shared__ alignas(16) unsigned short Bl[2][128 * 40];
  const int tid = threadIdx.x, lane = tid & 31, wid = tid >> 5;
  const int wm = wid >> 1, wn = wid & 1;
  const int M0 = blockIdx.y * 64, N0 = blockIdx.x * 128;

  v8f acc[4] = {};
#if HAVE_TDM
  if (wid == 0) {
    tdm_load_2d(ldsoff(Al[0]), Xbf + (size_t)M0 * 1024, 1024, 8192, 1024, 32, 64, 3, 3);
    tdm_load_2d(ldsoff(Bl[0]), Wt + (size_t)N0 * 1024, 1024, 3072, 1024, 32, 128, 3, 3);
  }
#endif
  for (int it = 0; it < 32; ++it) {
    const int cur = it & 1;
    __syncthreads();
#if HAVE_TDM
    if (wid == 0) {
      if (it + 1 < 32) {
        int kc2 = (it + 1) * 32;
        tdm_load_2d(ldsoff(Al[cur ^ 1]), Xbf + (size_t)M0 * 1024 + kc2, 1024, 8192, 1024, 32, 64, 3, 3);
        tdm_load_2d(ldsoff(Bl[cur ^ 1]), Wt + (size_t)N0 * 1024 + kc2, 1024, 3072, 1024, 32, 128, 3, 3);
        __builtin_amdgcn_s_wait_tensorcnt(2);
      } else {
        __builtin_amdgcn_s_wait_tensorcnt(0);
      }
    }
#else
    {
      int kc = it * 32;
      int r = tid >> 2, k0 = (tid & 3) * 8;
      *(uint4*)(Al[cur] + r * 40 + k0) = *(const uint4*)(Xbf + (size_t)(M0 + r) * 1024 + kc + k0);
      int rn = tid >> 1, kk = (tid & 1) * 16;
      *(uint4*)(Bl[cur] + rn * 40 + kk) = *(const uint4*)(Wt + (size_t)(N0 + rn) * 1024 + kc + kk);
      *(uint4*)(Bl[cur] + rn * 40 + kk + 8) = *(const uint4*)(Wt + (size_t)(N0 + rn) * 1024 + kc + kk + 8);
    }
#endif
    __syncthreads();
    v16bf a = load_a_frag(Al[cur], 40, wm * 16, 0, lane);
#pragma unroll
    for (int t = 0; t < 4; ++t) {
      v16bf b = load_b_frag(Bl[cur], 40, wn * 64 + 16 * t, 0, lane);
      acc[t] = WMMA_BF16(a, b, acc[t]);
    }
  }
  const int rowBase = M0 + wm * 16 + 8 * (lane >> 4);
#pragma unroll
  for (int t = 0; t < 4; ++t) {
    int c = N0 + wn * 64 + 16 * t + (lane & 15);
    float bias = bqkv[c];
    int which = c >> 10, rem = c & 1023, h = rem >> 6, d = rem & 63;
#pragma unroll
    for (int i = 0; i < 8; ++i) {
      int m = rowBase + i;
      int b = m >> 11, n = m & 2047, bh = b * 16 + h;
      unsigned short val = f2bf_raw(acc[t][i] + bias);
      if (which == 0)      Qws[((size_t)bh * 2048 + n) * 64 + d] = val;
      else if (which == 1) Kws[((size_t)bh * 2048 + n) * 64 + d] = val;
      else                 Vtws[((size_t)bh * 64 + d) * 2048 + n] = val;
    }
  }
}

// ---------------------------------------------------------------------------
// Kernel 2: flash attention. Block = (bh, 128 query rows), wave = 16 rows.
// TDM-fed, double-buffered 64-key K/V tiles; online softmax; O in registers.
// ---------------------------------------------------------------------------
__global__ __launch_bounds__(256) void attn_kernel(
    const unsigned short* __restrict__ Qws, const unsigned short* __restrict__ Kws,
    const unsigned short* __restrict__ Vtws, unsigned short* __restrict__ Ows) {
  constexpr int STR = 72;
  __shared__ alignas(16) unsigned short Kl[2][64 * STR];   // [key][d]
  __shared__ alignas(16) unsigned short Vl[2][64 * STR];   // [d][key]
  __shared__ alignas(16) unsigned short Pl[8 * 16 * STR];
  const int tid = threadIdx.x, lane = tid & 31, wv = tid >> 5;
  const int bh = blockIdx.y;
  const int qm0 = blockIdx.x * 128 + wv * 16;
  const unsigned short* Qb = Qws + (size_t)bh * 2048 * 64;
  const unsigned short* Kb = Kws + (size_t)bh * 2048 * 64;
  const unsigned short* Vb = Vtws + (size_t)bh * 64 * 2048;
  unsigned short* myP = Pl + wv * 16 * STR;

  v16bf qf0 = load_a_frag(Qb, 64, qm0, 0, lane);
  v16bf qf1 = load_a_frag(Qb, 64, qm0, 32, lane);

  float Mx[8], Ls[8];
  v8f o[4] = {};
#pragma unroll
  for (int i = 0; i < 8; ++i) { Mx[i] = -1e30f; Ls[i] = 0.0f; }

#if HAVE_TDM
  if (wv == 0) {
    tdm_load_2d(ldsoff(Kl[0]), Kb, 64, 2048, 64, 64, 64, 4, 3);
    tdm_load_2d(ldsoff(Vl[0]), Vb, 2048, 64, 2048, 64, 64, 4, 3);
  }
#endif
  for (int jb = 0; jb < 32; ++jb) {
    const int cur = jb & 1;
    const int j = jb * 64;
    __syncthreads();
#if HAVE_TDM
    if (wv == 0) {
      if (jb + 1 < 32) {
        tdm_load_2d(ldsoff(Kl[cur ^ 1]), Kb + (size_t)(j + 64) * 64, 64, 2048, 64, 64, 64, 4, 3);
        tdm_load_2d(ldsoff(Vl[cur ^ 1]), Vb + (j + 64), 2048, 64, 2048, 64, 64, 4, 3);
        __builtin_amdgcn_s_wait_tensorcnt(2);
      } else {
        __builtin_amdgcn_s_wait_tensorcnt(0);
      }
    }
#else
    {
      int r = tid >> 2, c0 = (tid & 3) * 16;
      *(uint4*)(Kl[cur] + r * STR + c0)     = *(const uint4*)(Kb + (size_t)(j + r) * 64 + c0);
      *(uint4*)(Kl[cur] + r * STR + c0 + 8) = *(const uint4*)(Kb + (size_t)(j + r) * 64 + c0 + 8);
      *(uint4*)(Vl[cur] + r * STR + c0)     = *(const uint4*)(Vb + (size_t)r * 2048 + j + c0);
      *(uint4*)(Vl[cur] + r * STR + c0 + 8) = *(const uint4*)(Vb + (size_t)r * 2048 + j + c0 + 8);
    }
#endif
    __syncthreads();

    v8f s[4] = {};
#pragma unroll
    for (int t = 0; t < 4; ++t) {
      v16bf bk0 = load_b_frag(Kl[cur], STR, 16 * t, 0, lane);
      v16bf bk1 = load_b_frag(Kl[cur], STR, 16 * t, 32, lane);
      s[t] = WMMA_BF16(qf0, bk0, s[t]);
      s[t] = WMMA_BF16(qf1, bk1, s[t]);
    }
#pragma unroll
    for (int t = 0; t < 4; ++t)
#pragma unroll
      for (int i = 0; i < 8; ++i) s[t][i] *= 0.125f;  // 1/sqrt(64)

    float corr[8], rsum[8];
#pragma unroll
    for (int i = 0; i < 8; ++i) {
      float v = fmaxf(fmaxf(s[0][i], s[1][i]), fmaxf(s[2][i], s[3][i]));
#pragma unroll
      for (int off = 1; off < 16; off <<= 1) v = fmaxf(v, __shfl_xor(v, off, 32));
      float nm = fmaxf(Mx[i], v);
      corr[i] = __expf(Mx[i] - nm);
      Mx[i] = nm;
      rsum[i] = 0.0f;
    }
#pragma unroll
    for (int t = 0; t < 4; ++t)
#pragma unroll
      for (int i = 0; i < 8; ++i) {
        float p = __expf(s[t][i] - Mx[i]);
        rsum[i] += p;
        myP[(i + 8 * (lane >> 4)) * STR + 16 * t + (lane & 15)] = f2bf_raw(p);
      }
#pragma unroll
    for (int i = 0; i < 8; ++i) {
#pragma unroll
      for (int off = 1; off < 16; off <<= 1) rsum[i] += __shfl_xor(rsum[i], off, 32);
      Ls[i] = Ls[i] * corr[i] + rsum[i];
    }
#pragma unroll
    for (int t = 0; t < 4; ++t)
#pragma unroll
      for (int i = 0; i < 8; ++i) o[t][i] *= corr[i];

    asm volatile("s_wait_dscnt 0" ::: "memory");  // P: C-layout -> A-layout staging
    v16bf pf0 = load_a_frag(myP, STR, 0, 0, lane);
    v16bf pf1 = load_a_frag(myP, STR, 0, 32, lane);
#pragma unroll
    for (int t = 0; t < 4; ++t) {
      v16bf bv0 = load_b_frag(Vl[cur], STR, 16 * t, 0, lane);
      v16bf bv1 = load_b_frag(Vl[cur], STR, 16 * t, 32, lane);
      o[t] = WMMA_BF16(pf0, bv0, o[t]);
      o[t] = WMMA_BF16(pf1, bv1, o[t]);
    }
  }

  float inv[8];
#pragma unroll
  for (int i = 0; i < 8; ++i) inv[i] = 1.0f / Ls[i];
  const int b = bh >> 4, h = bh & 15;
#pragma unroll
  for (int t = 0; t < 4; ++t) {
    int d = 16 * t + (lane & 15);
#pragma unroll
    for (int i = 0; i < 8; ++i) {
      int n = qm0 + i + 8 * (lane >> 4);
      Ows[((size_t)(b * 2048 + n)) * 1024 + h * 64 + d] = f2bf_raw(o[t][i] * inv[i]);
    }
  }
}

// ---------------------------------------------------------------------------
// Kernel 3: out = O @ WtP^T + b_proj (f32 out), TDM double-buffered tiles.
// ---------------------------------------------------------------------------
__global__ __launch_bounds__(256) void proj_gemm_kernel(
    const unsigned short* __restrict__ Ows, const unsigned short* __restrict__ Wt,
    const float* __restrict__ bp, float* __restrict__ out) {
  __shared__ alignas(16) unsigned short Al[2][64 * 40];
  __shared__ alignas(16) unsigned short Bl[2][128 * 40];
  const int tid = threadIdx.x, lane = tid & 31, wid = tid >> 5;
  const int wm = wid >> 1, wn = wid & 1;
  const int M0 = blockIdx.y * 64, N0 = blockIdx.x * 128;

  v8f acc[4] = {};
#if HAVE_TDM
  if (wid == 0) {
    tdm_load_2d(ldsoff(Al[0]), Ows + (size_t)M0 * 1024, 1024, 8192, 1024, 32, 64, 3, 3);
    tdm_load_2d(ldsoff(Bl[0]), Wt + (size_t)N0 * 1024, 1024, 1024, 1024, 32, 128, 3, 3);
  }
#endif
  for (int it = 0; it < 32; ++it) {
    const int cur = it & 1;
    __syncthreads();
#if HAVE_TDM
    if (wid == 0) {
      if (it + 1 < 32) {
        int kc2 = (it + 1) * 32;
        tdm_load_2d(ldsoff(Al[cur ^ 1]), Ows + (size_t)M0 * 1024 + kc2, 1024, 8192, 1024, 32, 64, 3, 3);
        tdm_load_2d(ldsoff(Bl[cur ^ 1]), Wt + (size_t)N0 * 1024 + kc2, 1024, 1024, 1024, 32, 128, 3, 3);
        __builtin_amdgcn_s_wait_tensorcnt(2);
      } else {
        __builtin_amdgcn_s_wait_tensorcnt(0);
      }
    }
#else
    {
      int kc = it * 32;
      int r = tid >> 2, k0 = (tid & 3) * 8;
      *(uint4*)(Al[cur] + r * 40 + k0) = *(const uint4*)(Ows + (size_t)(M0 + r) * 1024 + kc + k0);
      int rn = tid >> 1, kk = (tid & 1) * 16;
      *(uint4*)(Bl[cur] + rn * 40 + kk) = *(const uint4*)(Wt + (size_t)(N0 + rn) * 1024 + kc + kk);
      *(uint4*)(Bl[cur] + rn * 40 + kk + 8) = *(const uint4*)(Wt + (size_t)(N0 + rn) * 1024 + kc + kk + 8);
    }
#endif
    __syncthreads();
    v16bf a = load_a_frag(Al[cur], 40, wm * 16, 0, lane);
#pragma unroll
    for (int t = 0; t < 4; ++t) {
      v16bf b = load_b_frag(Bl[cur], 40, wn * 64 + 16 * t, 0, lane);
      acc[t] = WMMA_BF16(a, b, acc[t]);
    }
  }
  const int rowBase = M0 + wm * 16 + 8 * (lane >> 4);
#pragma unroll
  for (int t = 0; t < 4; ++t) {
    int c = N0 + wn * 64 + 16 * t + (lane & 15);
    float bias = bp[c];
#pragma unroll
    for (int i = 0; i < 8; ++i)
      out[(size_t)(rowBase + i) * 1024 + c] = acc[t][i] + bias;
  }
}

// ---------------------------------------------------------------------------
extern "C" void kernel_launch(void* const* d_in, const int* in_sizes, int n_in,
                              void* d_out, int out_size, void* d_ws, size_t ws_size,
                              hipStream_t stream) {
  (void)in_sizes; (void)n_in; (void)out_size; (void)ws_size;
  const float* x     = (const float*)d_in[0];
  const float* Wqkv  = (const float*)d_in[1];
  const float* bqkv  = (const float*)d_in[2];
  const float* Wproj = (const float*)d_in[3];
  const float* bproj = (const float*)d_in[4];
  float* out = (float*)d_out;

  const size_t seg = (size_t)64 * 2048 * 64;  // 8,388,608 bf16 elems
  unsigned short* Qws  = (unsigned short*)d_ws;
  unsigned short* Kws  = Qws + seg;
  unsigned short* Vtws = Kws + seg;
  unsigned short* Ows  = Vtws + seg;
  unsigned short* Xbf  = Ows + seg;
  unsigned short* WtQ  = Xbf + seg;                       // [3072][1024]
  unsigned short* WtP  = WtQ + (size_t)3072 * 1024;       // [1024][1024]

  cvt_bf16_kernel<<<4096, 256, 0, stream>>>(x, Xbf, 8192 * 1024);
  cvt_transpose_kernel<<<dim3(48, 16), 256, 0, stream>>>(Wqkv, WtQ, 3072);
  cvt_transpose_kernel<<<dim3(16, 16), 256, 0, stream>>>(Wproj, WtP, 1024);
  qkv_gemm_kernel<<<dim3(24, 128), 256, 0, stream>>>(Xbf, WtQ, bqkv, Qws, Kws, Vtws);
  attn_kernel<<<dim3(16, 64), 256, 0, stream>>>(Qws, Kws, Vtws, Ows);
  proj_gemm_kernel<<<dim3(8, 128), 256, 0, stream>>>(Ows, WtP, bproj, out);
}
